// SpectralConv2d_59992103191083
// MI455X (gfx1250) — compile-verified, matrix operand order checked by reading, not run on hardware
//
#include <hip/hip_runtime.h>
#include <math.h>

// ---------------------------------------------------------------------------
// CDNA5 WMMA plumbing (wave32, v_wmma_f32_16x16x32_bf16)
// ---------------------------------------------------------------------------
typedef __attribute__((ext_vector_type(16))) __bf16 v16bf;
typedef __attribute__((ext_vector_type(8)))  float  v8f;

struct alignas(16) V4u { unsigned int v[4]; };
struct V8u { V4u lo, hi; };  // 32 bytes, trivially copyable

// Native fptrunc -> hardware bf16 convert (v_cvt_pk_bf16_f32 on gfx1250).
// Scalar per-element form keeps dataflow explicit for the backend.
__device__ __forceinline__ unsigned short f2bf(float f) {
  __bf16 h = (__bf16)f;
  return __builtin_bit_cast(unsigned short, h);
}

// Load one 16x32 bf16 fragment. LDS tile stored K-contiguous with row stride
// ldk halves (row = M for A operand, row = N for a transposed-B operand).
// Per ISA 7.12.2: lane = {row: l&15, k-half: l>>4}; halves j0..7 -> K = 8*kh..,
// j8..15 -> K = 16+8*kh.. : two contiguous 16B runs -> two ds_load_b128.
__device__ __forceinline__ v16bf frag16x32(const unsigned short* p, int lane, int ldk) {
  const int r  = lane & 15;
  const int kh = lane >> 4;
  V8u t;
  t.lo = *(const V4u*)(p + r * ldk + 8 * kh);
  t.hi = *(const V4u*)(p + r * ldk + 16 + 8 * kh);
  return __builtin_bit_cast(v16bf, t);
}

__device__ __forceinline__ v8f wmma_bf16(v16bf a, v16bf b, v8f c) {
  // (neg_a, A, neg_b, B, c_mod, C, reuse_a, reuse_b) ; bf16 NEG[1:0] must be 0
  return __builtin_amdgcn_wmma_f32_16x16x32_bf16(false, a, false, b, (short)0, c,
                                                 false, false);
}

// ---------------------------------------------------------------------------
// Spectral axis kernel: per workgroup, 16 pencils of 64 positions x 64 ch.
//   F   = DFT_trunc(16x64) @ x(64x64)              (re/im)
//   O_m = F_m(16x64) @ W_m(64x64)  complex, 16 modes (batched over 16 pencils)
//   h  += iDFT(64x32) @ [Ore;Oim](32x64)
// x element offset = b*16777216 + c1*st1 + c2*st2 + pos*stp + i
// ---------------------------------------------------------------------------
#define SM_XT    0        // ushort[16][64][72]  x^T tiles   (73728)  | alias:
#define SM_OBUF  0        // ushort[16][64][32]  O spectra   (32768)  | (x dead)
#define SM_F     73728    // ushort[2][16][16][64] F spectra (32768)
#define SM_CRE   106496   // ushort[16][64]
#define SM_CIM   107520   // ushort[16][64]
#define SM_IDF   108544   // ushort[64][32]
#define SM_WRE   110592   // ushort[64][72]
#define SM_WIM   115200   // ushort[64][72]
#define SM_WNG   119808   // ushort[64][72]
#define SM_SPEC_BYTES (124416 * 2)   // 248832 B  (< 320 KB WGP LDS)

__global__ __launch_bounds__(256) void spectral_axis_kernel(
    const float* __restrict__ x, const float* __restrict__ w,
    float* __restrict__ h, int st1, int st2, int stp, int accumulate) {
  extern __shared__ unsigned short smem[];
  unsigned short* xT   = smem + SM_XT;
  unsigned short* Obuf = smem + SM_OBUF;
  unsigned short* F    = smem + SM_F;
  unsigned short* Cre  = smem + SM_CRE;
  unsigned short* Cim  = smem + SM_CIM;
  unsigned short* IDF  = smem + SM_IDF;
  unsigned short* WreT = smem + SM_WRE;
  unsigned short* WimT = smem + SM_WIM;
  unsigned short* WngT = smem + SM_WNG;

  const int tid  = threadIdx.x;
  const int lane = tid & 31;
  const int wave = tid >> 5;

  // ----- build ortho DFT / iDFT tables in bf16 (1/sqrt(64) = 1/8) -----
  const float W0 = 6.283185307179586f / 64.0f;
  for (int idx = tid; idx < 16 * 64; idx += 256) {
    int m = idx >> 6, z = idx & 63;
    float th = W0 * (float)(m * z);
    Cre[idx] = f2bf(cosf(th) * 0.125f);
    Cim[idx] = f2bf(-sinf(th) * 0.125f);          // e^{-i th}
  }
  for (int idx = tid; idx < 64 * 32; idx += 256) {
    int z = idx >> 5, kk = idx & 31, m = kk & 15;
    float th = W0 * (float)(m * z);
    float s  = (m == 0 ? 1.0f : 2.0f) * 0.125f;   // hermitian doubling
    IDF[idx] = f2bf(kk < 16 ? s * cosf(th) : -s * sinf(th));
  }

  // ----- stage 16 pencils of x into LDS, bf16, transposed to [i][pos] -----
  const int slab0 = blockIdx.x * 16;
  for (int idx = tid; idx < 16 * 64 * 64; idx += 256) {
    int s = idx >> 12, pos = (idx >> 6) & 63, i = idx & 63;
    int sg = slab0 + s;
    int b = sg >> 12, rr = sg & 4095, c1 = rr >> 6, c2 = rr & 63;
    long base = (long)b * 16777216 + (long)c1 * st1 + (long)c2 * st2;
    xT[(s * 64 + i) * 72 + pos] = f2bf(x[base + (long)pos * stp + i]);
  }
  __syncthreads();

  // ----- stage 1: F[reim][m][s][i] = C @ x   (256 WMMA / WG) -----
  for (int u = wave; u < 128; u += 8) {
    int s = u >> 3, reim = (u >> 2) & 1, nt = u & 3;
    const unsigned short* C = reim ? Cim : Cre;
    v8f acc = {};
    for (int kt = 0; kt < 2; ++kt) {
      v16bf a = frag16x32(C + kt * 32, lane, 64);
      v16bf b = frag16x32(xT + (s * 64 + nt * 16) * 72 + kt * 32, lane, 72);
      acc = wmma_bf16(a, b, acc);
    }
    int n = lane & 15, kh = lane >> 4;
    for (int r = 0; r < 8; ++r) {
      int m = r + 8 * kh;
      F[((reim * 16 + m) * 16 + s) * 64 + nt * 16 + n] = f2bf(acc[r]);
    }
  }
  __syncthreads();

  // ----- stage 2: per-mode complex channel mix (512 WMMA / WG) -----
  for (int m = 0; m < 16; ++m) {
    // stage W_m^T (re, im, -im) ; weights are L2-resident (1.5 MB total)
    for (int idx = tid; idx < 4096; idx += 256) {
      int o = idx & 63, i = idx >> 6;
      const float* wp = w + (((long)i * 64 + o) * 16 + m) * 2;
      float re = wp[0], im = wp[1];
      WreT[o * 72 + i] = f2bf(re);
      WimT[o * 72 + i] = f2bf(im);
      WngT[o * 72 + i] = f2bf(-im);
    }
    if (m < 15)  // warm L2->L0 path for next mode (global_prefetch_b8)
      __builtin_prefetch(w + (((long)(tid & 63) * 64) * 16 + (m + 1)) * 2, 0, 1);
    __syncthreads();

    // 8 waves = 4 o-tiles x {Ore, Oim}
    {
      int nt = wave & 3, ro = wave >> 2;
      const unsigned short* B0 = ro ? WimT : WreT;  // pairs with Fre
      const unsigned short* B1 = ro ? WreT : WngT;  // pairs with Fim
      v8f acc = {};
      for (int kt = 0; kt < 2; ++kt) {
        v16bf a = frag16x32(F + ((0 * 16 + m) * 16) * 64 + kt * 32, lane, 64);
        v16bf b = frag16x32(B0 + nt * 16 * 72 + kt * 32, lane, 72);
        acc = wmma_bf16(a, b, acc);
      }
      for (int kt = 0; kt < 2; ++kt) {
        v16bf a = frag16x32(F + ((1 * 16 + m) * 16) * 64 + kt * 32, lane, 64);
        v16bf b = frag16x32(B1 + nt * 16 * 72 + kt * 32, lane, 72);
        acc = wmma_bf16(a, b, acc);
      }
      int n = lane & 15, kh = lane >> 4;
      for (int r = 0; r < 8; ++r) {
        int s = r + 8 * kh;
        Obuf[(s * 64 + nt * 16 + n) * 32 + ro * 16 + m] = f2bf(acc[r]);
      }
    }
    __syncthreads();
  }

  // ----- stage 3: inverse DFT, K=32 in one WMMA step (256 WMMA / WG) -----
  for (int u = wave; u < 256; u += 8) {
    int s = u >> 4, mt = (u >> 2) & 3, nt = u & 3;
    v16bf a = frag16x32(IDF + mt * 16 * 32, lane, 32);
    v16bf b = frag16x32(Obuf + (s * 64 + nt * 16) * 32, lane, 32);
    v8f acc = {};
    acc = wmma_bf16(a, b, acc);

    int sg = slab0 + s;
    int bb = sg >> 12, rr = sg & 4095, c1 = rr >> 6, c2 = rr & 63;
    long base = (long)bb * 16777216 + (long)c1 * st1 + (long)c2 * st2;
    int n = lane & 15, kh = lane >> 4;
    for (int r = 0; r < 8; ++r) {
      int z = mt * 16 + r + 8 * kh;
      long addr = base + (long)z * stp + nt * 16 + n;
      float v = acc[r];
      if (accumulate) v += h[addr];
      h[addr] = v;
    }
  }
}

// ---------------------------------------------------------------------------
// FFN kernel: out = relu(h @ W1 + b1) @ W2 + b2, run twice (backcast/forecast)
// 128-row tile per WG; weights transposed into bf16 LDS (L2-resident).
// ---------------------------------------------------------------------------
#define FM_XT   0                    // ushort[128][72]
#define FM_W1T  9216                 // ushort[256][72]
#define FM_W2T  27648                // ushort[64][264]
#define FM_HT   44544                // ushort[128][264]
#define FM_END  78336                // ushorts
#define SM_FFN_BYTES (FM_END * 2 + 320 * 4)   // + b1[256], b2[64] floats

__global__ __launch_bounds__(256) void ffn_kernel(
    const float* __restrict__ h,
    const float* __restrict__ w1a, const float* __restrict__ b1a,
    const float* __restrict__ w2a, const float* __restrict__ b2a,
    const float* __restrict__ w1b, const float* __restrict__ b1b,
    const float* __restrict__ w2b, const float* __restrict__ b2b,
    float* __restrict__ outB, float* __restrict__ outF) {
  extern __shared__ unsigned short smem[];
  unsigned short* Xt  = smem + FM_XT;
  unsigned short* W1T = smem + FM_W1T;
  unsigned short* W2T = smem + FM_W2T;
  unsigned short* HT  = smem + FM_HT;
  float* b1 = (float*)(smem + FM_END);
  float* b2 = b1 + 256;

  const int tid = threadIdx.x, lane = tid & 31, wave = tid >> 5;
  const long row0 = (long)blockIdx.x * 128;

  // X tile (shared by both FFNs)
  for (int idx = tid; idx < 128 * 64; idx += 256) {
    int rrow = idx >> 6, i = idx & 63;
    Xt[rrow * 72 + i] = f2bf(h[(row0 + rrow) * 64 + i]);
  }

  for (int ff = 0; ff < 2; ++ff) {
    const float* w1 = ff ? w1b : w1a;  const float* bb1 = ff ? b1b : b1a;
    const float* w2 = ff ? w2b : w2a;  const float* bb2 = ff ? b2b : b2a;
    float* out = ff ? outF : outB;

    __syncthreads();  // Xt ready / previous FFN done with W,HT
    for (int idx = tid; idx < 64 * 256; idx += 256) {   // W1 (64,256) -> W1T
      int n = idx & 255, k = idx >> 8;
      W1T[n * 72 + k] = f2bf(w1[k * 256 + n]);
    }
    for (int idx = tid; idx < 256 * 64; idx += 256) {   // W2 (256,64) -> W2T
      int o = idx & 63, k = idx >> 6;
      W2T[o * 264 + k] = f2bf(w2[k * 64 + o]);
    }
    b1[tid] = bb1[tid];
    if (tid < 64) b2[tid] = bb2[tid];
    __syncthreads();

    // GEMM1 + bias + ReLU -> HT   (256 WMMA / WG)
    for (int u = wave; u < 128; u += 8) {
      int mt = u >> 4, nt = u & 15;
      v8f acc = {};
      for (int kt = 0; kt < 2; ++kt) {
        v16bf a = frag16x32(Xt + mt * 16 * 72 + kt * 32, lane, 72);
        v16bf b = frag16x32(W1T + nt * 16 * 72 + kt * 32, lane, 72);
        acc = wmma_bf16(a, b, acc);
      }
      int n = lane & 15, kh = lane >> 4;
      float bias = b1[nt * 16 + n];
      for (int r = 0; r < 8; ++r) {
        int rrow = mt * 16 + r + 8 * kh;
        float v = acc[r] + bias;
        HT[rrow * 264 + nt * 16 + n] = f2bf(v > 0.0f ? v : 0.0f);
      }
    }
    __syncthreads();

    // GEMM2 + bias -> out (fp32)   (256 WMMA / WG)
    for (int u = wave; u < 32; u += 8) {
      int mt = u >> 2, nt = u & 3;
      v8f acc = {};
      for (int kt = 0; kt < 8; ++kt) {
        v16bf a = frag16x32(HT + mt * 16 * 264 + kt * 32, lane, 264);
        v16bf b = frag16x32(W2T + nt * 16 * 264 + kt * 32, lane, 264);
        acc = wmma_bf16(a, b, acc);
      }
      int n = lane & 15, kh = lane >> 4;
      for (int r = 0; r < 8; ++r) {
        int rrow = mt * 16 + r + 8 * kh, col = nt * 16 + n;
        out[(row0 + rrow) * 64 + col] = acc[r] + b2[col];
      }
    }
  }
}

// ---------------------------------------------------------------------------
extern "C" void kernel_launch(void* const* d_in, const int* in_sizes, int n_in,
                              void* d_out, int out_size, void* d_ws, size_t ws_size,
                              hipStream_t stream) {
  const float* x    = (const float*)d_in[0];
  const float* wx   = (const float*)d_in[1];
  const float* wy   = (const float*)d_in[2];
  const float* wz   = (const float*)d_in[3];
  const float* b_w1 = (const float*)d_in[4];
  const float* b_b1 = (const float*)d_in[5];
  const float* b_w2 = (const float*)d_in[6];
  const float* b_b2 = (const float*)d_in[7];
  const float* f_w1 = (const float*)d_in[8];
  const float* f_b1 = (const float*)d_in[9];
  const float* f_w2 = (const float*)d_in[10];
  const float* f_b2 = (const float*)d_in[11];

  float* h    = (float*)d_ws;                 // 2*64^3*64 fp32 = 134 MB
  float* outB = (float*)d_out;
  float* outF = outB + (size_t)33554432;      // tuple concat: (backcast, forecast)

  dim3 blk(256);
  dim3 gspec(512);   // 8192 pencils / 16 per WG

  // element offset(b,x,y,z,i) = b*16777216 + x*262144 + y*4096 + z*64 + i
  // z-axis pass (writes h)
  spectral_axis_kernel<<<gspec, blk, SM_SPEC_BYTES, stream>>>(
      x, wz, h, 262144, 4096, 64, 0);
  // y-axis pass (accumulates)
  spectral_axis_kernel<<<gspec, blk, SM_SPEC_BYTES, stream>>>(
      x, wy, h, 262144, 64, 4096, 1);
  // x-axis pass (accumulates)
  spectral_axis_kernel<<<gspec, blk, SM_SPEC_BYTES, stream>>>(
      x, wx, h, 4096, 64, 262144, 1);

  // FFNs: 524288 rows / 128 per WG
  ffn_kernel<<<dim3(4096), blk, SM_FFN_BYTES, stream>>>(
      h, b_w1, b_b1, b_w2, b_b2, f_w1, f_b1, f_w2, f_b2, outB, outF);
}
